// interactionModule_nonParametric_torque_28269474742524
// MI455X (gfx1250) — matrix operand order repaired and checked by reading, not compiled, not run on hardware
//
#include <hip/hip_runtime.h>
#include <hip/hip_bf16.h>

typedef __attribute__((ext_vector_type(16))) _Float16 v16h;
typedef __attribute__((ext_vector_type(8)))  _Float16 v8h;
typedef __attribute__((ext_vector_type(8)))  float    v8f;

#define HD 128   // hidden width

// ---------------------------------------------------------------------------
// Pack W1/W2 (fp32 [128][128], row = input K, col = output N) into a
// WMMA B-fragment layout, f16, contiguous W1 frags then W2 frags (64 KB total)
// so the edge kernel can async-copy one linear region into LDS.
// B 32x16 f16 layout used here (contiguous k, same as A): lane L holds column n = n_tile*16 + (L&15),
// K values k0..k0+15 with k0 = k_step*32 + (L>>4)*16.
// ---------------------------------------------------------------------------
__global__ void pack_weights_kernel(const float* __restrict__ W1,
                                    const float* __restrict__ W2,
                                    _Float16* __restrict__ Wpacked) {
    const int frag  = blockIdx.x;          // 0..63  (layer*32 + f)
    const int lane  = threadIdx.x;         // 0..31
    const int layer = frag >> 5;
    const int f     = frag & 31;
    const int kstep = f >> 3;
    const int ntile = f & 7;

    const float* W = layer ? W2 : W1;

    const int n  = ntile * 16 + (lane & 15);
    const int k0 = kstep * 32 + (lane >> 4) * 16;

    _Float16* out = Wpacked + ((size_t)(frag * 32 + lane)) * 16;
#pragma unroll
    for (int j = 0; j < 16; ++j)
        out[j] = (_Float16)W[(size_t)(k0 + j) * HD + n];
}

__global__ void zero_f32_kernel(float* __restrict__ p, int n) {
    int i = blockIdx.x * blockDim.x + threadIdx.x;
    if (i < n) p[i] = 0.0f;
}

// ---------------------------------------------------------------------------
// Edge MLP. 256 threads = 8 waves; each wave processes 16 edges.
// Block prologue: async-copy 64 KB of packed weights into LDS
// (global_load_async_to_lds_b128, ASYNCcnt) overlapped with the feature +
// layer-0 phases. Per wave: feat[16][4] -> h0[16][128] (f16, LDS) ->
// 2x (16x128)@(128x128) via v_wmma_f32_16x16x32_f16 fed from LDS ->
// final 128-dot -> atomicAdd into per-dst accumulators.
// ---------------------------------------------------------------------------
__global__ __launch_bounds__(256, 2)
void edge_mlp_kernel(const float* __restrict__ x,
                     const float* __restrict__ theta,
                     const int*   __restrict__ src,
                     const int*   __restrict__ dst,
                     const float* __restrict__ W0,
                     const float* __restrict__ b0,
                     const _Float16* __restrict__ Wpacked,  // 32768 halves
                     const float* __restrict__ b1,
                     const float* __restrict__ b2,
                     const float* __restrict__ W3,
                     const float* __restrict__ b3,
                     float* __restrict__ sums,
                     float* __restrict__ cnt,
                     int E) {
    __shared__ __attribute__((aligned(32))) _Float16 lds_w[2 * 32 * 32 * 16]; // 64 KB
    __shared__ __attribute__((aligned(32))) _Float16 lds_h[8][16][HD];        // 32 KB
    __shared__ __attribute__((aligned(16))) float    lds_feat[8][16][4];      //  2 KB

    const int lane = threadIdx.x & 31;
    const int wave = threadIdx.x >> 5;
    const int row  = lane & 15;   // edge-in-wave / C column index
    const int hi   = lane >> 4;   // half-wave id
    const int ebase = blockIdx.x * 128 + wave * 16;

    // ---- kick off async weight staging: 64 KB global -> LDS ----
    {
        const unsigned tid      = threadIdx.x;
        const unsigned lds_base = (unsigned)(uintptr_t)(&lds_w[0]);
        const char*    g        = (const char*)Wpacked;
#pragma unroll
        for (int it = 0; it < 16; ++it) {
            unsigned           off   = (unsigned)it * 4096u + tid * 16u;
            unsigned           laddr = lds_base + off;
            unsigned long long gaddr = (unsigned long long)(uintptr_t)(g + off);
            asm volatile("global_load_async_to_lds_b128 %0, %1, off"
                         :: "v"(laddr), "v"(gaddr) : "memory");
        }
    }

    // ---- per-edge geometric features (lanes 0..15 of each wave) ----
    if (hi == 0) {
        int e = ebase + row;
        int ec = e < E ? e : (E - 1);           // clamp OOB reads
        int s = src[ec], d = dst[ec];
        float xs0 = x[2 * s], xs1 = x[2 * s + 1];
        float xd0 = x[2 * d], xd1 = x[2 * d + 1];
        float ts = theta[s], td = theta[d];
        float dr0 = xd0 - xs0, dr1 = xd1 - xs1;
        float dth = td - ts;
        float cs = __cosf(ts), ss = __sinf(ts);
        lds_feat[wave][row][0] =  dr0 * cs + dr1 * ss;   // dx in src frame
        lds_feat[wave][row][1] = -dr0 * ss + dr1 * cs;   // dy in src frame
        lds_feat[wave][row][2] = __cosf(dth);
        lds_feat[wave][row][3] = __sinf(dth);
    }
    asm volatile("s_wait_dscnt 0x0" ::: "memory");

    // ---- layer 0: feat[16x4] @ W0[4x128] (+b0, ReLU) -> lds_h (f16) ----
    {
        float f0 = lds_feat[wave][row][0];
        float f1 = lds_feat[wave][row][1];
        float f2 = lds_feat[wave][row][2];
        float f3 = lds_feat[wave][row][3];
#pragma unroll 8
        for (int i = 0; i < 64; ++i) {
            int n = hi * 64 + i;
            float v = b0[n] + f0 * W0[n] + f1 * W0[HD + n]
                            + f2 * W0[2 * HD + n] + f3 * W0[3 * HD + n];
            lds_h[wave][row][n] = (_Float16)fmaxf(v, 0.0f);
        }
    }
    asm volatile("s_wait_dscnt 0x0" ::: "memory");

    // ---- weights must be resident in LDS for all waves past this point ----
    asm volatile("s_wait_asynccnt 0x0" ::: "memory");
    __syncthreads();

    // ---- layers 1 & 2: 16x128 @ 128x128 with v_wmma_f32_16x16x32_f16 ----
#pragma unroll 1
    for (int layer = 0; layer < 2; ++layer) {
        const _Float16* wl   = lds_w + layer * 16384;
        const float*    bias = layer ? b2 : b1;

        v8f c[8];
#pragma unroll
        for (int t = 0; t < 8; ++t) c[t] = (v8f){};

#pragma unroll
        for (int ks = 0; ks < 4; ++ks) {
            // A fragment (16x32 f16): lane<16 holds K {k+0..15},
            // lane>=16 holds K {k+16..31} of its row.
            const int kb = ks * 32 + hi * 16;
            v8h alo = *(const v8h*)&lds_h[wave][row][kb];
            v8h ahi = *(const v8h*)&lds_h[wave][row][kb + 8];
            v16h a;
#pragma unroll
            for (int j = 0; j < 8; ++j) { a[j] = alo[j]; a[8 + j] = ahi[j]; }

#pragma unroll
            for (int nt = 0; nt < 8; ++nt) {
                v16h b = *(const v16h*)(wl + ((size_t)((ks * 8 + nt) * 32 + lane)) * 16);
                c[nt] = __builtin_amdgcn_wmma_f32_16x16x32_f16(
                            false, a, false, b, (short)0, c[nt], false, false);
            }
        }

        // bias + ReLU + f16 store back to the wave-private activation tile.
        // C layout: column n = nt*16 + (lane&15); VGPR r holds M = r + hi*8.
#pragma unroll
        for (int nt = 0; nt < 8; ++nt) {
            int n = nt * 16 + row;
            float bv = bias[n];
#pragma unroll
            for (int r = 0; r < 8; ++r) {
                float v = fmaxf(c[nt][r] + bv, 0.0f);
                lds_h[wave][r + hi * 8][n] = (_Float16)v;
            }
        }
        asm volatile("s_wait_dscnt 0x0" ::: "memory");
    }

    // ---- layer 3: h2[16x128] @ W3[128x1] + b3, aggregate into dst ----
    if (hi == 0) {
        int e = ebase + row;
        if (e < E) {
            float acc = b3[0];
#pragma unroll 16
            for (int n = 0; n < HD; ++n)
                acc += (float)lds_h[wave][row][n] * W3[n];
            int d = dst[e];
            atomicAdd(&sums[d], acc);
            atomicAdd(&cnt[d], 1.0f);
        }
    }
}

// ---------------------------------------------------------------------------
// Finalize: velocity = v0*[cos th, sin th] ; torque = sum / max(cnt, 1)
// d_out layout: velocity [N,2] flat (2N floats) then torque [N,1] (N floats).
// ---------------------------------------------------------------------------
__global__ void finalize_kernel(const float* __restrict__ theta,
                                const float* __restrict__ v0,
                                const float* __restrict__ sums,
                                const float* __restrict__ cnt,
                                float* __restrict__ out, int N) {
    int i = blockIdx.x * blockDim.x + threadIdx.x;
    if (i < N) {
        float t = theta[i];
        float v = v0[0];
        out[2 * i]     = v * __cosf(t);
        out[2 * i + 1] = v * __sinf(t);
        out[2 * N + i] = sums[i] / fmaxf(cnt[i], 1.0f);
    }
}

extern "C" void kernel_launch(void* const* d_in, const int* in_sizes, int n_in,
                              void* d_out, int out_size, void* d_ws, size_t ws_size,
                              hipStream_t stream) {
    (void)n_in; (void)out_size; (void)ws_size;
    const float* x     = (const float*)d_in[0];
    const float* theta = (const float*)d_in[1];
    const float* v0    = (const float*)d_in[2];
    const float* W0    = (const float*)d_in[3];
    const float* b0    = (const float*)d_in[4];
    const float* W1    = (const float*)d_in[5];
    const float* b1    = (const float*)d_in[6];
    const float* W2    = (const float*)d_in[7];
    const float* b2    = (const float*)d_in[8];
    const float* W3    = (const float*)d_in[9];
    const float* b3    = (const float*)d_in[10];
    const int*   src   = (const int*)d_in[11];
    const int*   dst   = (const int*)d_in[12];

    const int N = in_sizes[1];        // theta is [N,1]
    const int E = in_sizes[11];       // src is [E]

    // workspace layout: packed W1|W2 frags (64KB) | sums[N] | cnt[N]
    char* ws = (char*)d_ws;
    _Float16* Wpacked = (_Float16*)ws;
    float*    sums    = (float*)(ws + 65536);
    float*    cntb    = sums + N;

    pack_weights_kernel<<<64, 32, 0, stream>>>(W1, W2, Wpacked);
    zero_f32_kernel<<<(2 * N + 255) / 256, 256, 0, stream>>>(sums, 2 * N);

    const int blocks = (E + 127) / 128;
    edge_mlp_kernel<<<blocks, 256, 0, stream>>>(
        x, theta, src, dst, W0, b0, Wpacked, b1, b2, W3, b3, sums, cntb, E);

    finalize_kernel<<<(N + 255) / 256, 256, 0, stream>>>(theta, v0, sums, cntb,
                                                         (float*)d_out, N);
}
